// MultiHeadSelfAttention_24696061952273
// MI455X (gfx1250) — compile-verified, matrix operand order checked by reading, not run on hardware
//
#include <hip/hip_runtime.h>
#include <stdint.h>

#define BATCH  4
#define SEQ    2048
#define DMODEL 1024
#define NHEADS 16
#define DK     64

// Detect gfx1250 async global->LDS builtins (toolchain-dependent arity/names;
// fall back to load+ds_store which keeps identical semantics and LDS reuse).
#if defined(__has_builtin)
#if __has_builtin(__builtin_amdgcn_global_load_async_to_lds_b128)
#define HAVE_ASYNC_LDS 1
#endif
#endif
#ifndef HAVE_ASYNC_LDS
#define HAVE_ASYNC_LDS 0
#endif

typedef __attribute__((ext_vector_type(16))) __bf16 v16bf;
typedef __attribute__((ext_vector_type(8)))  __bf16 bf16x8;
typedef __attribute__((ext_vector_type(8)))  float  v8f;

// Exact pointee type the async-to-LDS builtin expects (per hipcc diagnostic):
// 'int __attribute__((vector_size(16)))' in addrspace(1)/(3).
typedef int vi4 __attribute__((vector_size(4 * sizeof(int))));
typedef __attribute__((address_space(1))) vi4 gvi4;
typedef __attribute__((address_space(3))) vi4 lvi4;

// Load one WMMA 16-bit operand's per-lane data: two contiguous 8-element
// chunks at p[0..7] and p[16..23] (caller bakes in the half*8 lane swizzle).
__device__ __forceinline__ v16bf load_op(const __bf16* p) {
  bf16x8 lo = *(const bf16x8*)(p);
  bf16x8 hi = *(const bf16x8*)(p + 16);
  v16bf r;
#pragma unroll
  for (int i = 0; i < 8; ++i) { r[i] = lo[i]; r[i + 8] = hi[i]; }
  return r;
}

__device__ __forceinline__ v8f wmma_bf16(v16bf a, v16bf b, v8f c) {
  return __builtin_amdgcn_wmma_f32_16x16x32_bf16(
      /*neg_a=*/false, a, /*neg_b=*/false, b,
      /*c_mod=*/(short)0, c, /*reuse_a=*/false, /*reuse_b=*/false);
}

__device__ __forceinline__ void wait_async0() {
#if defined(__has_builtin) && __has_builtin(__builtin_amdgcn_s_wait_asynccnt)
  __builtin_amdgcn_s_wait_asynccnt(0);
#else
  asm volatile("s_wait_asynccnt 0x0" ::: "memory");
#endif
}

// ---------------------------------------------------------------- fp32->bf16
__global__ void cvt_f32_bf16(const float* __restrict__ src,
                             __bf16* __restrict__ dst, int n) {
  int i = blockIdx.x * blockDim.x + threadIdx.x;
  int stride = gridDim.x * blockDim.x;
  for (; i < n; i += stride) dst[i] = (__bf16)src[i];
}

// ------------------------------------------------------------------- GEMM
// Y[m,n] = sum_k A[m,k] * W[n,k]   (A: [M,K] row-major, W: [N,K] row-major)
// MODE 0: out fp32 [M,N]
// MODE 1: out bf16 [B,H,S,DK]   (Q, K)
// MODE 2: out bf16 [B,H,DK,S]   (V transposed for PV B-operand)
// Block = 256 threads (8 waves), block tile 128x64, wave tile 32x32.
// The 64x32 weight tile is staged into LDS per k-step (async-to-LDS when the
// toolchain exposes it) and shared by the 4 waves per column group (4x less
// L2 weight traffic); A operands stream straight from global into WMMA regs.
template <int MODE>
__global__ __launch_bounds__(256)
void gemm_bf16(const __bf16* __restrict__ A,
               const __bf16* __restrict__ W,
               void* __restrict__ outp,
               int M, int N, int K) {
  __shared__ __align__(16) __bf16 wtile[64][32];

  const int tid  = threadIdx.x;
  const int lane = tid & 31;
  const int wave = tid >> 5;
  const int half = lane >> 4;
  const int l16  = lane & 15;
  const int m0   = blockIdx.y * 128 + (wave & 3) * 32;
  const int nblk = blockIdx.x * 64;

  // staging: each of 256 threads moves one 16B chunk of the 64x32 bf16 tile
  const int srow = tid >> 2;          // 0..63
  const int schk = (tid & 3) * 8;     // 0,8,16,24
  const __bf16* wsrc = W + (size_t)(nblk + srow) * K + schk;
  __bf16*       wdst = &wtile[srow][schk];

  v8f acc00 = {}, acc01 = {}, acc10 = {}, acc11 = {};

  const __bf16* a0p = A + (size_t)(m0 + l16) * K + half * 8;
  const __bf16* a1p = a0p + (size_t)16 * K;
  const __bf16* b0row = &wtile[(wave >> 2) * 32 + l16][half * 8];
  const __bf16* b1row = b0row + 16 * 32;

  for (int k = 0; k < K; k += 32) {
    __syncthreads();                       // previous tile fully consumed
#if HAVE_ASYNC_LDS
    {
      __builtin_amdgcn_global_load_async_to_lds_b128(
          (gvi4*)(uintptr_t)(const void*)(wsrc + k),
          (lvi4*)(uint32_t)(uintptr_t)(void*)wdst,
          /*offset=*/0, /*cpol=*/0);
      wait_async0();
    }
#else
    *(bf16x8*)wdst = *(const bf16x8*)(wsrc + k);
#endif
    __syncthreads();                       // tile visible to all waves

    v16bf a0 = load_op(a0p + k);
    v16bf a1 = load_op(a1p + k);
    v16bf b0 = load_op(b0row);             // ds_load_b128 x2
    v16bf b1 = load_op(b1row);
    __builtin_prefetch(a0p + k + 128, 0, 3);
    acc00 = wmma_bf16(a0, b0, acc00);
    acc01 = wmma_bf16(a0, b1, acc01);
    acc10 = wmma_bf16(a1, b0, acc10);
    acc11 = wmma_bf16(a1, b1, acc11);
  }

  const int n0 = nblk + (wave >> 2) * 32;
  v8f accs[2][2] = {{acc00, acc01}, {acc10, acc11}};
#pragma unroll
  for (int ti = 0; ti < 2; ++ti) {
#pragma unroll
    for (int tj = 0; tj < 2; ++tj) {
#pragma unroll
      for (int r = 0; r < 8; ++r) {
        // C/D layout: VGPR r, lanes 0-15 -> M=r, lanes 16-31 -> M=r+8
        int m = m0 + ti * 16 + half * 8 + r;
        int n = n0 + tj * 16 + l16;
        float v = accs[ti][tj][r];
        if (MODE == 0) {
          ((float*)outp)[(size_t)m * N + n] = v;
        } else {
          int b = m / SEQ, s = m % SEQ;
          int h = n / DK,  d = n % DK;
          __bf16* o = (__bf16*)outp;
          if (MODE == 1)
            o[(((size_t)(b * NHEADS + h) * SEQ) + s) * DK + d] = (__bf16)v;
          else
            o[(((size_t)(b * NHEADS + h) * DK) + d) * SEQ + s] = (__bf16)v;
        }
      }
    }
  }
}

// ------------------------------------------------------------ flash attention
// Q,K: [B,H,S,DK] bf16; Vt: [B,H,DK,S] bf16; Ob: [B,S,H*DK] bf16.
// Block = 128 threads (4 waves); each wave owns a 16-query tile, iterates
// causal key range in steps of 32 with online softmax; QK^T and PV via WMMA.
__global__ __launch_bounds__(128)
void flash_attn(const __bf16* __restrict__ Q,
                const __bf16* __restrict__ Km,
                const __bf16* __restrict__ Vt,
                __bf16* __restrict__ Ob) {
  __shared__ __align__(16) __bf16 pbuf[4][16][32];  // per-wave P staging

  const int lane = threadIdx.x & 31;
  const int wave = threadIdx.x >> 5;
  const int half = lane >> 4;
  const int l16  = lane & 15;
  const int bh = blockIdx.y;               // b*H + h
  const int b  = bh / NHEADS;
  const int h  = bh % NHEADS;
  const int q0 = blockIdx.x * 64 + wave * 16;

  const __bf16* Qbase = Q  + ((size_t)bh * SEQ + q0) * DK;
  const __bf16* Kbase = Km + (size_t)bh * SEQ * DK;
  const __bf16* Vbase = Vt + (size_t)bh * DK * SEQ;

  // Q A-operands (loop invariant): rows = l16, dk chunks per lane swizzle
  v16bf qa0 = load_op(Qbase + (size_t)l16 * DK + half * 8);       // dk 0..31
  v16bf qa1 = load_op(Qbase + (size_t)l16 * DK + 32 + half * 8);  // dk 32..63

  v8f o0 = {}, o1 = {}, o2 = {}, o3 = {};
  float ms[8], ls[8];
#pragma unroll
  for (int r = 0; r < 8; ++r) { ms[r] = -3.0e38f; ls[r] = 0.0f; }

  const float scale = 0.125f;  // 1/sqrt(64)
  const int kend = q0 + 16;    // causal: keys < q0+16 can be unmasked

  for (int kt = 0; kt < kend; kt += 32) {
    // ---- scores for key subtiles [kt,kt+16) and [kt+16,kt+32)
    v8f s0 = {}, s1 = {};
    {
      const __bf16* kp0 = Kbase + (size_t)(kt + l16) * DK + half * 8;
      v16bf kb0 = load_op(kp0);
      v16bf kb1 = load_op(kp0 + 32);
      s0 = wmma_bf16(qa0, kb0, s0);
      s0 = wmma_bf16(qa1, kb1, s0);
      const __bf16* kp1 = kp0 + (size_t)16 * DK;
      v16bf kc0 = load_op(kp1);
      v16bf kc1 = load_op(kp1 + 32);
      s1 = wmma_bf16(qa0, kc0, s1);
      s1 = wmma_bf16(qa1, kc1, s1);
    }

    // ---- causal mask + online softmax (rows live in 16-lane halves)
    float alpha[8];
#pragma unroll
    for (int r = 0; r < 8; ++r) {
      int m = r + half * 8;
      int qidx = q0 + m;
      float v0 = s0[r] * scale;
      float v1 = s1[r] * scale;
      if (kt + l16 > qidx)      v0 = -3.0e38f;
      if (kt + 16 + l16 > qidx) v1 = -3.0e38f;
      float mx = fmaxf(v0, v1);
#pragma unroll
      for (int msk = 1; msk < 16; msk <<= 1)
        mx = fmaxf(mx, __shfl_xor(mx, msk, 32));
      float mnew = fmaxf(ms[r], mx);
      float a  = __expf(ms[r] - mnew);
      float p0 = __expf(v0 - mnew);
      float p1 = __expf(v1 - mnew);
      float rs = p0 + p1;
#pragma unroll
      for (int msk = 1; msk < 16; msk <<= 1)
        rs += __shfl_xor(rs, msk, 32);
      ls[r] = ls[r] * a + rs;
      ms[r] = mnew;
      alpha[r] = a;
      // stage P (C-layout) into LDS for A-layout re-read
      pbuf[wave][m][l16]      = (__bf16)p0;
      pbuf[wave][m][l16 + 16] = (__bf16)p1;
    }

    // rescale running output by alpha
#pragma unroll
    for (int r = 0; r < 8; ++r) {
      o0[r] *= alpha[r]; o1[r] *= alpha[r];
      o2[r] *= alpha[r]; o3[r] *= alpha[r];
    }

    __builtin_amdgcn_wave_barrier();
    // P as WMMA A-operand: row = l16, key chunks half*8 / half*8+16
    v16bf pa = load_op(&pbuf[wave][l16][half * 8]);
    __builtin_amdgcn_wave_barrier();

    // ---- PV: (16 queries x 32 keys) x (32 keys x 64 dk), Vt key-contiguous
    {
      const __bf16* vp = Vbase + (size_t)l16 * SEQ + kt + half * 8;
      o0 = wmma_bf16(pa, load_op(vp + (size_t)0 * 16 * SEQ), o0);
      o1 = wmma_bf16(pa, load_op(vp + (size_t)1 * 16 * SEQ), o1);
      o2 = wmma_bf16(pa, load_op(vp + (size_t)2 * 16 * SEQ), o2);
      o3 = wmma_bf16(pa, load_op(vp + (size_t)3 * 16 * SEQ), o3);
    }
  }

  // ---- epilogue: normalize and store bf16 [B,S,H*DK]
#pragma unroll
  for (int r = 0; r < 8; ++r) {
    float inv = 1.0f / ls[r];
    int m = r + half * 8;
    size_t row = ((size_t)b * SEQ + q0 + m) * DMODEL + (size_t)h * DK;
    Ob[row + 0  + l16] = (__bf16)(o0[r] * inv);
    Ob[row + 16 + l16] = (__bf16)(o1[r] * inv);
    Ob[row + 32 + l16] = (__bf16)(o2[r] * inv);
    Ob[row + 48 + l16] = (__bf16)(o3[r] * inv);
  }
}

// ------------------------------------------------------------------ launch
extern "C" void kernel_launch(void* const* d_in, const int* in_sizes, int n_in,
                              void* d_out, int out_size, void* d_ws, size_t ws_size,
                              hipStream_t stream) {
  const float* x  = (const float*)d_in[0];
  const float* wq = (const float*)d_in[1];
  const float* wk = (const float*)d_in[2];
  const float* wv = (const float*)d_in[3];
  const float* wo = (const float*)d_in[4];

  const size_t MTOK = (size_t)BATCH * SEQ;          // 8192 token rows
  const size_t XE   = MTOK * DMODEL;                // activation elements
  const size_t WE   = (size_t)DMODEL * DMODEL;      // weight elements

  __bf16* p   = (__bf16*)d_ws;
  __bf16* xb  = p; p += XE;
  __bf16* wqb = p; p += WE;
  __bf16* wkb = p; p += WE;
  __bf16* wvb = p; p += WE;
  __bf16* wob = p; p += WE;
  __bf16* Qb  = p; p += XE;   // [B,H,S,DK]
  __bf16* Kb  = p; p += XE;   // [B,H,S,DK]
  __bf16* Vtb = p; p += XE;   // [B,H,DK,S]
  __bf16* Ab  = p; p += XE;   // attention output bf16 [B,S,D]

  cvt_f32_bf16<<<1024, 256, 0, stream>>>(x,  xb,  (int)XE);
  cvt_f32_bf16<<<512,  256, 0, stream>>>(wq, wqb, (int)WE);
  cvt_f32_bf16<<<512,  256, 0, stream>>>(wk, wkb, (int)WE);
  cvt_f32_bf16<<<512,  256, 0, stream>>>(wv, wvb, (int)WE);
  cvt_f32_bf16<<<512,  256, 0, stream>>>(wo, wob, (int)WE);

  dim3 gb(DMODEL / 64, (unsigned)(MTOK / 128));   // (16, 64)
  gemm_bf16<1><<<gb, 256, 0, stream>>>(xb, wqb, Qb,  (int)MTOK, DMODEL, DMODEL);
  gemm_bf16<1><<<gb, 256, 0, stream>>>(xb, wkb, Kb,  (int)MTOK, DMODEL, DMODEL);
  gemm_bf16<2><<<gb, 256, 0, stream>>>(xb, wvb, Vtb, (int)MTOK, DMODEL, DMODEL);

  dim3 ga(SEQ / 64, BATCH * NHEADS);              // (32, 64)
  flash_attn<<<ga, 128, 0, stream>>>(Qb, Kb, Vtb, Ab);

  gemm_bf16<0><<<gb, 256, 0, stream>>>(Ab, wob, d_out, (int)MTOK, DMODEL, DMODEL);
}
